// MultiHeadAttention_57629871177981
// MI455X (gfx1250) — matrix operand order repaired and checked by reading, not compiled, not run on hardware
//
#include <hip/hip_runtime.h>

// ---------------------------------------------------------------------------
// MI455X (gfx1250) multi-head attention, bf16 WMMA everywhere, fp32 accum.
// B=4, S=2048, D=1024, H=16, HD=64.
// ---------------------------------------------------------------------------

typedef __attribute__((ext_vector_type(16))) __bf16 v16bf;
typedef __attribute__((ext_vector_type(8)))  float  v8f;

constexpr int Bb = 4, Hh = 16, Ss = 2048, Dd = 1024, HD = 64;
constexpr int BM = 128, BN = 64, BK = 32;     // projection GEMM workgroup tile
constexpr float EPSV = 1e-7f;

union V16U { v16bf v; uint4 q[2]; };

__device__ __forceinline__ v8f wmma_bf16(v16bf a, v16bf b, v8f c) {
  // v_wmma_f32_16x16x32_bf16 : D = A(16x32) * B(32x16) + C(16x16 f32)
  return __builtin_amdgcn_wmma_f32_16x16x32_bf16(
      /*neg_a=*/false, a, /*neg_b=*/false, b,
      /*c_mod=*/(short)0, c, /*reuse_a=*/false, /*reuse_b=*/false);
}

// A-fragment (16x32 bf16): lane L holds row M=L&15.
// elements 0..7  -> K = 8*(L>=16) + 0..7
// elements 8..15 -> K = 16 + 8*(L>=16) + 0..7
__device__ __forceinline__ v16bf ld_afrag(const __bf16* row, int hi) {
  V16U u;
  u.q[0] = *(const uint4*)(row + hi * 8);
  u.q[1] = *(const uint4*)(row + 16 + hi * 8);
  return u.v;
}
// B-fragment (32x16 bf16), from a row holding all 32 K-values contiguously
// for column N = lane&15: elements 0..7 -> K = 8*(L>=16) + 0..7,
// elements 8..15 -> K = 16 + 8*(L>=16) + 0..7.
__device__ __forceinline__ v16bf ld_bfrag(const __bf16* row, int hi) {
  V16U u;
  u.q[0] = *(const uint4*)(row + hi * 8);
  u.q[1] = *(const uint4*)(row + 16 + hi * 8);
  return u.v;
}

__device__ __forceinline__ unsigned pk2(float a, float b) {
  union { __bf16 h; unsigned short u; } x, y;
  x.h = (__bf16)a; y.h = (__bf16)b;
  return (unsigned)x.u | ((unsigned)y.u << 16);
}

// ---------------------------------------------------------------------------
// GEMM: Out = relu(A[8192x1024] * W[1024x1024] + bias), bf16 WMMA, f32 accum.
// IN_BF16: A is bf16 (attention context) vs fp32 (x input)
// SPLIT:   write bf16 [B,H,S,HD] (projection) vs fp32 [M,N] (final output)
// ---------------------------------------------------------------------------
template <bool IN_BF16, bool SPLIT>
__global__ __launch_bounds__(256) void gemm_relu_kernel(
    const void* __restrict__ Ain, const float* __restrict__ W,
    const float* __restrict__ bias, void* __restrict__ Out) {
  __shared__ __bf16 xs[BM][BK + 8];   // 128 x 40 bf16 (A tile)
  __shared__ __bf16 wt[BN][BK + 8];   // 64 x 40 bf16 (W tile, transposed: [col][k])

  const int t    = threadIdx.x;
  const int lane = t & 31, wave = t >> 5;
  const int wr   = wave & 3;          // 4 wave-rows of 32
  const int wc   = wave >> 2;         // 2 wave-cols of 32
  const int hi   = lane >> 4, ln = lane & 15;
  const int row0 = blockIdx.x * BM;
  const int col0 = blockIdx.y * BN;

  v8f c00 = {}, c01 = {}, c10 = {}, c11 = {};

  for (int k0 = 0; k0 < Dd; k0 += BK) {
    // ---- stage A tile (128x32) as bf16 ----
    {
      const int r = t >> 1;
      const int cb = (t & 1) * 16;
      if (!IN_BF16) {
        const float* src = (const float*)Ain + (size_t)(row0 + r) * Dd + k0 + cb;
        float4 f0 = ((const float4*)src)[0];
        float4 f1 = ((const float4*)src)[1];
        float4 f2 = ((const float4*)src)[2];
        float4 f3 = ((const float4*)src)[3];
        uint4 u0 = { pk2(f0.x, f0.y), pk2(f0.z, f0.w), pk2(f1.x, f1.y), pk2(f1.z, f1.w) };
        uint4 u1 = { pk2(f2.x, f2.y), pk2(f2.z, f2.w), pk2(f3.x, f3.y), pk2(f3.z, f3.w) };
        *(uint4*)&xs[r][cb]     = u0;
        *(uint4*)&xs[r][cb + 8] = u1;
      } else {
        const __bf16* src = (const __bf16*)Ain + (size_t)(row0 + r) * Dd + k0 + cb;
        *(uint4*)&xs[r][cb]     = ((const uint4*)src)[0];
        *(uint4*)&xs[r][cb + 8] = ((const uint4*)src)[1];
      }
    }
    // ---- stage W tile (32x64) transposed -> wt[col][k] ----
    {
      const int r = t >> 3;           // k-local 0..31
      const int cb = (t & 7) * 8;     // col-local 0..56
      const float* src = W + (size_t)(k0 + r) * Dd + col0 + cb;
      float4 f0 = ((const float4*)src)[0];
      float4 f1 = ((const float4*)src)[1];
      wt[cb + 0][r] = (__bf16)f0.x; wt[cb + 1][r] = (__bf16)f0.y;
      wt[cb + 2][r] = (__bf16)f0.z; wt[cb + 3][r] = (__bf16)f0.w;
      wt[cb + 4][r] = (__bf16)f1.x; wt[cb + 5][r] = (__bf16)f1.y;
      wt[cb + 6][r] = (__bf16)f1.z; wt[cb + 7][r] = (__bf16)f1.w;
      if (k0 + BK < Dd) {
        __builtin_prefetch(W + (size_t)(k0 + BK + r) * Dd + col0 + cb, 0, 3);
      }
    }
    __syncthreads();

    const int ar = wr * 32 + ln;
    v16bf a0 = ld_afrag(&xs[ar][0], hi);
    v16bf a1 = ld_afrag(&xs[ar + 16][0], hi);
    const int bc = wc * 32 + ln;
    v16bf b0 = ld_bfrag(&wt[bc][0], hi);
    v16bf b1 = ld_bfrag(&wt[bc + 16][0], hi);

    c00 = wmma_bf16(a0, b0, c00);
    c01 = wmma_bf16(a0, b1, c01);
    c10 = wmma_bf16(a1, b0, c10);
    c11 = wmma_bf16(a1, b1, c11);
    __syncthreads();
  }

  // ---- epilogue: bias + relu, store ----
  const int n0 = col0 + wc * 32 + ln;
  const int n1 = n0 + 16;
  const float bv0 = bias[n0];
  const float bv1 = bias[n1];
#pragma unroll
  for (int v = 0; v < 8; ++v) {
    const int m0 = row0 + wr * 32 + v + 8 * hi;
    const int m1 = m0 + 16;
    float r00 = fmaxf(c00[v] + bv0, 0.f);
    float r01 = fmaxf(c01[v] + bv1, 0.f);
    float r10 = fmaxf(c10[v] + bv0, 0.f);
    float r11 = fmaxf(c11[v] + bv1, 0.f);
    if (SPLIT) {
      __bf16* o = (__bf16*)Out;
      // [B,H,S,HD]: m = b*S+s ; n = h*HD+d
      o[((size_t)((m0 >> 11) * Hh + (n0 >> 6)) * Ss + (m0 & (Ss - 1))) * HD + (n0 & 63)] = (__bf16)r00;
      o[((size_t)((m0 >> 11) * Hh + (n1 >> 6)) * Ss + (m0 & (Ss - 1))) * HD + (n1 & 63)] = (__bf16)r01;
      o[((size_t)((m1 >> 11) * Hh + (n0 >> 6)) * Ss + (m1 & (Ss - 1))) * HD + (n0 & 63)] = (__bf16)r10;
      o[((size_t)((m1 >> 11) * Hh + (n1 >> 6)) * Ss + (m1 & (Ss - 1))) * HD + (n1 & 63)] = (__bf16)r11;
    } else {
      float* o = (float*)Out;
      o[(size_t)m0 * Dd + n0] = r00;
      o[(size_t)m0 * Dd + n1] = r01;
      o[(size_t)m1 * Dd + n0] = r10;
      o[(size_t)m1 * Dd + n1] = r11;
    }
  }
}

// ---------------------------------------------------------------------------
// Attention: per WG = one (b,h) x 128 queries; each wave owns 16 queries.
// Flash-style single pass; row max taken over ALL keys (reference subtracts
// the pre-mask max), multiplicative causal mask applied after exp, +eps in
// the denominator. Output: bf16 merged-head layout [B,S,D].
// ---------------------------------------------------------------------------
__global__ __launch_bounds__(256) void attn_kernel(
    const __bf16* __restrict__ Qg, const __bf16* __restrict__ Kg,
    const __bf16* __restrict__ Vg, __bf16* __restrict__ Ctx) {
  __shared__ __bf16 Ks[32][72];        // key chunk  [key][hd]
  __shared__ __bf16 Vts[64][40];       // value chunk transposed [hd][key]
  __shared__ float  sc[8][16][36];     // per-wave score tile [q][k]
  __shared__ __bf16 pb[8][16][40];     // per-wave p tile (bf16) [q][k]
  __shared__ float  abuf[8][16];       // per-wave alpha broadcast
  __shared__ float  lbuf[8][16];       // per-wave 1/(l+eps) broadcast

  const int t = threadIdx.x, lane = t & 31, wave = t >> 5;
  const int hi = lane >> 4, ln = lane & 15;
  const int bh = blockIdx.y;
  const int qw = blockIdx.x * 128 + wave * 16;   // this wave's first query
  const __bf16* Qb = Qg + (size_t)bh * Ss * HD;
  const __bf16* Kb = Kg + (size_t)bh * Ss * HD;
  const __bf16* Vb = Vg + (size_t)bh * Ss * HD;

  // Q A-fragments for hd chunks [0,32) and [32,64), scaled by 1/sqrt(64).
  v16bf qa0, qa1;
  {
    const __bf16* qrow = Qb + (size_t)(qw + ln) * HD;
#pragma unroll
    for (int e = 0; e < 16; ++e) {
      int kk = (e & 7) + ((e >> 3) << 4) + hi * 8;
      qa0[e] = (__bf16)(0.125f * (float)qrow[kk]);
      qa1[e] = (__bf16)(0.125f * (float)qrow[32 + kk]);
    }
  }

  v8f o0 = {}, o1 = {}, o2 = {}, o3 = {};   // ctx accum: hd blocks of 16
  float mrow = -1e30f, lrow = 0.f;          // row state (row = ln, both halves)
  const int qg = qw + ln;                   // this row's query index

  const int r8 = t >> 3;          // staging: key-local row 0..31
  const int c8 = (t & 7) * 8;     // staging: hd column 0..56

  for (int kb = 0; kb < Ss; kb += 32) {
    const bool live = (kb <= qw + 15);      // wave-uniform: any causal key here
    __syncthreads();              // protect Ks/Vts from previous iteration
    // ---- stage K chunk via CDNA5 async global->LDS DMA (ASYNCcnt) ----
    {
      unsigned ldsK = (unsigned)(size_t)&Ks[r8][c8];
      const __bf16* gK = Kb + (size_t)(kb + r8) * HD + c8;
      asm volatile("global_load_async_to_lds_b128 %0, %1, off"
                   :: "v"(ldsK), "v"(gK)
                   : "memory");
      // ---- stage V chunk transposed through VGPRs ----
      uint4 dv = *(const uint4*)(Vb + (size_t)(kb + r8) * HD + c8);
      const __bf16* pv = (const __bf16*)&dv;
#pragma unroll
      for (int j = 0; j < 8; ++j) Vts[c8 + j][r8] = pv[j];
      if (kb + 32 < Ss) {
        __builtin_prefetch(Kb + (size_t)(kb + 32 + r8) * HD + c8, 0, 3);
        __builtin_prefetch(Vb + (size_t)(kb + 32 + r8) * HD + c8, 0, 3);
      }
      asm volatile("s_wait_asynccnt 0" ::: "memory");
    }
    __syncthreads();

    // ---- scores S[16q x 32k] = Qs @ K^T  (always: max is over ALL keys) ----
    v8f s0 = {}, s1 = {};
    {
      v16bf k00 = ld_bfrag(&Ks[ln][0], hi);        // keys kb+ln,    hd 0..31
      v16bf k01 = ld_bfrag(&Ks[ln][32], hi);       // keys kb+ln,    hd 32..63
      v16bf k10 = ld_bfrag(&Ks[ln + 16][0], hi);   // keys kb+16+ln, hd 0..31
      v16bf k11 = ld_bfrag(&Ks[ln + 16][32], hi);
      s0 = wmma_bf16(qa0, k00, s0); s0 = wmma_bf16(qa1, k01, s0);
      s1 = wmma_bf16(qa0, k10, s1); s1 = wmma_bf16(qa1, k11, s1);
    }
#pragma unroll
    for (int v = 0; v < 8; ++v) {
      sc[wave][v + 8 * hi][ln]      = s0[v];
      sc[wave][v + 8 * hi][16 + ln] = s1[v];
    }
    __syncthreads();

    // ---- softmax bookkeeping: lane L owns (row=ln, key half=hi) ----
    {
      const float* srow = &sc[wave][ln][hi * 16];
      float s16[16];
      float mx = mrow;
#pragma unroll
      for (int j = 0; j < 16; j += 4) {
        float4 f = *(const float4*)(srow + j);
        s16[j] = f.x; s16[j + 1] = f.y; s16[j + 2] = f.z; s16[j + 3] = f.w;
        mx = fmaxf(mx, fmaxf(fmaxf(f.x, f.y), fmaxf(f.z, f.w)));
      }
      mx = fmaxf(mx, __shfl_xor(mx, 16, 32));   // combine key halves
      float alpha = __expf(mrow - mx);
      mrow = mx;
      float addl = 0.f;
      if (live) {
        const int kbase = kb + hi * 16;
        unsigned pk[8];
        float sum16 = 0.f;
#pragma unroll
        for (int j = 0; j < 16; j += 2) {
          float p0 = (kbase + j     <= qg) ? __expf(s16[j]     - mx) : 0.f;
          float p1 = (kbase + j + 1 <= qg) ? __expf(s16[j + 1] - mx) : 0.f;
          sum16 += p0 + p1;
          pk[j >> 1] = pk2(p0, p1);
        }
        uint4* pdst = (uint4*)&pb[wave][ln][hi * 16];
        pdst[0] = *(uint4*)&pk[0];
        pdst[1] = *(uint4*)&pk[4];
        addl = sum16 + __shfl_xor(sum16, 16, 32);
      }
      lrow = lrow * alpha + addl;
      if (lane < 16) abuf[wave][ln] = alpha;
    }
    __syncthreads();

    // ---- rescale accumulators (max may grow even on masked chunks) ----
#pragma unroll
    for (int v = 0; v < 8; ++v) {
      float a = abuf[wave][v + 8 * hi];
      o0[v] *= a; o1[v] *= a; o2[v] *= a; o3[v] *= a;
    }

    // ---- ctx += p @ V (skip chunks fully in the future for this wave) ----
    if (live) {
      v16bf pa  = ld_afrag(&pb[wave][ln][0], hi);
      v16bf vb0 = ld_bfrag(&Vts[ln][0], hi);
      v16bf vb1 = ld_bfrag(&Vts[16 + ln][0], hi);
      v16bf vb2 = ld_bfrag(&Vts[32 + ln][0], hi);
      v16bf vb3 = ld_bfrag(&Vts[48 + ln][0], hi);
      o0 = wmma_bf16(pa, vb0, o0);
      o1 = wmma_bf16(pa, vb1, o1);
      o2 = wmma_bf16(pa, vb2, o2);
      o3 = wmma_bf16(pa, vb3, o3);
    }
  }

  // ---- normalize (denominator + eps) and store merged-head bf16 ctx ----
  if (lane < 16) lbuf[wave][ln] = 1.f / (lrow + EPSV);
  __syncthreads();
  const int b = bh >> 4, h = bh & 15;
#pragma unroll
  for (int v = 0; v < 8; ++v) {
    float linv = lbuf[wave][v + 8 * hi];
    int s = qw + v + 8 * hi;
    size_t ro = ((size_t)b * Ss + s) * Dd + h * HD;
    Ctx[ro + ln]      = (__bf16)(o0[v] * linv);
    Ctx[ro + 16 + ln] = (__bf16)(o1[v] * linv);
    Ctx[ro + 32 + ln] = (__bf16)(o2[v] * linv);
    Ctx[ro + 48 + ln] = (__bf16)(o3[v] * linv);
  }
}

// ---------------------------------------------------------------------------
extern "C" void kernel_launch(void* const* d_in, const int* in_sizes, int n_in,
                              void* d_out, int out_size, void* d_ws, size_t ws_size,
                              hipStream_t stream) {
  (void)in_sizes; (void)n_in; (void)out_size; (void)ws_size;
  const float* x  = (const float*)d_in[0];
  const float* Wq = (const float*)d_in[1]; const float* bq = (const float*)d_in[2];
  const float* Wk = (const float*)d_in[3]; const float* bk = (const float*)d_in[4];
  const float* Wv = (const float*)d_in[5]; const float* bv = (const float*)d_in[6];
  const float* Wo = (const float*)d_in[7]; const float* bo = (const float*)d_in[8];

  char* ws = (char*)d_ws;
  const size_t SZ = (size_t)Bb * Hh * Ss * HD * sizeof(__bf16);   // 16 MB each
  __bf16* Qw = (__bf16*)(ws);
  __bf16* Kw = (__bf16*)(ws + SZ);
  __bf16* Vw = (__bf16*)(ws + 2 * SZ);
  __bf16* Cw = (__bf16*)(ws + 3 * SZ);

  dim3 gG((Bb * Ss) / BM, Dd / BN);          // 64 x 16
  gemm_relu_kernel<false, true><<<gG, 256, 0, stream>>>(x, Wq, bq, Qw);
  gemm_relu_kernel<false, true><<<gG, 256, 0, stream>>>(x, Wk, bk, Kw);
  gemm_relu_kernel<false, true><<<gG, 256, 0, stream>>>(x, Wv, bv, Vw);

  attn_kernel<<<dim3(Ss / 128, Bb * Hh), 256, 0, stream>>>(Qw, Kw, Vw, Cw);

  gemm_relu_kernel<true, false><<<gG, 256, 0, stream>>>(Cw, Wo, bo, d_out);
}
